// SparsePertokenMoE_5772436046048
// MI455X (gfx1250) — compile-verified, matrix operand order checked by reading, not run on hardware
//
#include <hip/hip_runtime.h>

// ---------------------------------------------------------------------------
// Sparse per-token MoE (top-1 routed expert + shared expert), bf16 WMMA path
// for MI455X / gfx1250.
//   DIM=1024, HID=4096, NUM_EXPERTS=8 (7 real), TOP_K=2 but loop uses top-1,
//   ALPHA=2.0, tokens = 4*2048 = 8192.
// Tile: 32 tokens x 1024 out per block, 512 threads = 16 waves. Each weight
// B-fragment feeds 2 WMMAs (two stacked M-tiles) -> ~1 b128 load per WMMA.
// ---------------------------------------------------------------------------

#define DIM   1024
#define HID   4096
#define NEXP  8
#define NREAL 7
#define NTOK  8192
#define ALPHA 2.0f
#define TM    32          // tokens per workgroup tile

typedef __attribute__((ext_vector_type(16))) __bf16 v16bf;
typedef __attribute__((ext_vector_type(8)))  float  v8f;

union Frag16 {            // 16 bf16 = 32 bytes = 8 VGPRs
    v16bf v;
    uint4 u[2];
};

static __device__ __forceinline__ v8f wmma_bf16(v16bf a, v16bf b, v8f c) {
    // D = A(16x32 bf16) x B(32x16 bf16) + C(16x16 f32)
    return __builtin_amdgcn_wmma_f32_16x16x32_bf16(
        /*neg_a=*/false, a, /*neg_b=*/false, b,
        /*c_mod=*/(short)0, c, /*reuse_a=*/false, /*reuse_b=*/false);
}

// ---------------------------------------------------------------------------
// 1) fp32 -> bf16 copy (activations)
// ---------------------------------------------------------------------------
__global__ __launch_bounds__(256) void cvt_bf16_kernel(const float* __restrict__ in,
                                                       __bf16* __restrict__ out, int n) {
    int i = blockIdx.x * blockDim.x + threadIdx.x;
    int stride = gridDim.x * blockDim.x;
    for (; i < n; i += stride) out[i] = (__bf16)in[i];
}

// ---------------------------------------------------------------------------
// 2) fp32 RxC -> bf16 CxR tiled transpose (weights), one matrix per blockIdx.z
// ---------------------------------------------------------------------------
__global__ __launch_bounds__(256) void transpose_bf16_kernel(const float* __restrict__ in,
                                                             __bf16* __restrict__ outT,
                                                             int R, int C) {
    __shared__ float tile[32][33];
    const float* inp  = in   + (size_t)blockIdx.z * R * C;
    __bf16*      outp = outT + (size_t)blockIdx.z * R * C;
    int c0 = blockIdx.x * 32, r0 = blockIdx.y * 32;
    #pragma unroll
    for (int i = threadIdx.y; i < 32; i += 8) {
        tile[i][threadIdx.x] = inp[(size_t)(r0 + i) * C + (c0 + threadIdx.x)];
    }
    __syncthreads();
    #pragma unroll
    for (int i = threadIdx.y; i < 32; i += 8) {
        outp[(size_t)(c0 + i) * R + (r0 + threadIdx.x)] = (__bf16)tile[threadIdx.x][i];
    }
}

// ---------------------------------------------------------------------------
// 3) Router: one wave32 per token -> softmax top-1, scatter to expert lists
// ---------------------------------------------------------------------------
__global__ __launch_bounds__(256) void router_kernel(const float* __restrict__ x,
                                                     const float* __restrict__ Wr,
                                                     float* __restrict__ pvals,
                                                     int* __restrict__ counts,
                                                     int* __restrict__ lists) {
    int gtid = blockIdx.x * blockDim.x + threadIdx.x;
    int tok  = gtid >> 5;
    int lane = threadIdx.x & 31;
    if (tok >= NTOK) return;
    const float* xr = x + (size_t)tok * DIM;
    float acc[NEXP];
    #pragma unroll
    for (int e = 0; e < NEXP; ++e) acc[e] = 0.f;
    for (int k = lane; k < DIM; k += 32) {
        float xv = xr[k];
        const float* wr = Wr + (size_t)k * NEXP;
        #pragma unroll
        for (int e = 0; e < NEXP; ++e) acc[e] += xv * wr[e];
    }
    #pragma unroll
    for (int off = 16; off > 0; off >>= 1) {
        #pragma unroll
        for (int e = 0; e < NEXP; ++e) acc[e] += __shfl_xor(acc[e], off, 32);
    }
    if (lane == 0) {
        float m = acc[0]; int bi = 0;
        #pragma unroll
        for (int e = 1; e < NEXP; ++e) { if (acc[e] > m) { m = acc[e]; bi = e; } }
        float s = 0.f;
        #pragma unroll
        for (int e = 0; e < NEXP; ++e) s += __expf(acc[e] - m);
        pvals[tok] = ALPHA / s;            // ALPHA * softmax top-1 (exp(0)/s)
        if (bi < NREAL) {
            int pos = atomicAdd(&counts[bi], 1);
            lists[(size_t)bi * NTOK + pos] = tok;
        }
    }
}

// ---------------------------------------------------------------------------
// Core 32-token SwiGLU tile: out[32 x 1024] (+)= scale * (silu(x@Wg)*(x@Wu))@Wd
// 512 threads = 16 waves. Weights are pre-transposed bf16:
//   WuT/WgT: [HID][DIM]  (row n = column n of Wu/Wg)
//   WdT    : [DIM][HID]  (row n = column n of Wd)
// Phase A: wave w computes up+gate for 16 hidden cols (2 M-tiles).
// Phase B: wave w accumulates 64 output cols x 32 rows.
// ---------------------------------------------------------------------------
#define XS_STRIDE 1032   // 1024 + 8 bf16 pad (2064 B, 16B multiple)
#define HS_STRIDE 264    // 256 + 8 bf16 pad  (528 B, 16B multiple)
#define HCHUNK    256    // hidden columns per chunk (16 chunks of 256)

template <bool GATHER>
static __device__ void swiglu_tile(const __bf16* __restrict__ xb,
                                   const __bf16* __restrict__ WuT,
                                   const __bf16* __restrict__ WgT,
                                   const __bf16* __restrict__ WdT,
                                   float* __restrict__ out,
                                   const int* __restrict__ toklist,
                                   const float* __restrict__ pvals,
                                   int row_base, int valid) {
    __shared__ __align__(16) __bf16 xs[TM * XS_STRIDE];
    __shared__ __align__(16) __bf16 hs[TM * HS_STRIDE];
    __shared__ int   tok_s[TM];
    __shared__ float scl_s[TM];

    const int tid  = threadIdx.x;
    const int w    = tid >> 5;          // wave id 0..15
    const int lane = tid & 31;
    const int rsel = lane & 15;         // row (A) / col (B) selector
    const int kvar = (lane >> 4) << 3;  // 0 or 8: K sub-block per ISA layout
    const int mhi  = (lane >> 4) << 3;  // C/D: M = r + mhi (+ 16*mt)

    if (tid < TM) {
        int tok; float s;
        if (GATHER) {
            tok = toklist[(tid < valid) ? tid : 0];
            s   = (tid < valid) ? pvals[tok] : 0.f;
        } else {
            tok = row_base + tid;
            s   = 1.f;
        }
        tok_s[tid] = tok;
        scl_s[tid] = s;
    }
    __syncthreads();

    // Stage x tile [32 x 1024] bf16 into LDS (16B chunks).
    for (int i = tid; i < TM * 128; i += 512) {
        int r  = i >> 7;
        int cc = (i & 127) << 3;  // element offset, 8 bf16 per chunk
        uint4 v = *(const uint4*)(xb + (size_t)tok_s[r] * DIM + cc);
        *(uint4*)(xs + r * XS_STRIDE + cc) = v;
    }
    __syncthreads();

    v8f acc_out[2][4] = {};   // [mt][nt]: 32 rows x 64 cols (cols w*64..)

    for (int hc = 0; hc < HID / HCHUNK; ++hc) {
        // ---- Phase A: up & gate for this wave's 16 hidden columns ----
        const int colbase = (hc << 8) + (w << 4);
        v8f au[2] = {}, ag[2] = {};     // [mt]
        const __bf16* wu0 = WuT + (size_t)(colbase + rsel) * DIM;
        const __bf16* wg0 = WgT + (size_t)(colbase + rsel) * DIM;

        for (int kc = 0; kc < DIM; kc += 32) {
            Frag16 a0, a1;
            a0.u[0] = *(const uint4*)(xs + rsel * XS_STRIDE + kc + kvar);
            a0.u[1] = *(const uint4*)(xs + rsel * XS_STRIDE + kc + 16 + kvar);
            a1.u[0] = *(const uint4*)(xs + (16 + rsel) * XS_STRIDE + kc + kvar);
            a1.u[1] = *(const uint4*)(xs + (16 + rsel) * XS_STRIDE + kc + 16 + kvar);
            if (kc + 32 < DIM) {
                __builtin_prefetch(wu0 + kc + 32, 0, 1);
                __builtin_prefetch(wg0 + kc + 32, 0, 1);
            }
            Frag16 b;
            b.u[0] = *(const uint4*)(wu0 + kc + kvar);
            b.u[1] = *(const uint4*)(wu0 + kc + 16 + kvar);
            au[0] = wmma_bf16(a0.v, b.v, au[0]);
            au[1] = wmma_bf16(a1.v, b.v, au[1]);
            b.u[0] = *(const uint4*)(wg0 + kc + kvar);
            b.u[1] = *(const uint4*)(wg0 + kc + 16 + kvar);
            ag[0] = wmma_bf16(a0.v, b.v, ag[0]);
            ag[1] = wmma_bf16(a1.v, b.v, ag[1]);
        }

        // silu(g)*up -> bf16 h chunk in LDS
        #pragma unroll
        for (int mt = 0; mt < 2; ++mt) {
            #pragma unroll
            for (int r = 0; r < 8; ++r) {
                float gv = ag[mt][r];
                float hv = au[mt][r] * (gv / (1.f + __expf(-gv)));
                int M = (mt << 4) + r + mhi;
                hs[M * HS_STRIDE + (w << 4) + rsel] = (__bf16)hv;
            }
        }
        __syncthreads();

        // ---- Phase B: down-projection, this wave owns 64 output columns ----
        for (int kc = 0; kc < HCHUNK; kc += 32) {
            Frag16 a0, a1;
            a0.u[0] = *(const uint4*)(hs + rsel * HS_STRIDE + kc + kvar);
            a0.u[1] = *(const uint4*)(hs + rsel * HS_STRIDE + kc + 16 + kvar);
            a1.u[0] = *(const uint4*)(hs + (16 + rsel) * HS_STRIDE + kc + kvar);
            a1.u[1] = *(const uint4*)(hs + (16 + rsel) * HS_STRIDE + kc + 16 + kvar);
            #pragma unroll
            for (int nt = 0; nt < 4; ++nt) {
                const __bf16* wd = WdT + (size_t)((w << 6) + (nt << 4) + rsel) * HID
                                       + (hc << 8) + kc;
                Frag16 b;
                b.u[0] = *(const uint4*)(wd + kvar);
                b.u[1] = *(const uint4*)(wd + 16 + kvar);
                acc_out[0][nt] = wmma_bf16(a0.v, b.v, acc_out[0][nt]);
                acc_out[1][nt] = wmma_bf16(a1.v, b.v, acc_out[1][nt]);
            }
        }
        __syncthreads();   // hs reused next chunk
    }

    // ---- Writeback ----
    #pragma unroll
    for (int mt = 0; mt < 2; ++mt) {
        #pragma unroll
        for (int nt = 0; nt < 4; ++nt) {
            #pragma unroll
            for (int r = 0; r < 8; ++r) {
                int M = (mt << 4) + r + mhi;
                int col = (w << 6) + (nt << 4) + rsel;
                if (GATHER) {
                    if (M < valid) {
                        size_t o = (size_t)tok_s[M] * DIM + col;
                        out[o] += scl_s[M] * acc_out[mt][nt][r];  // rows disjoint
                    }
                } else {
                    out[(size_t)tok_s[M] * DIM + col] = acc_out[mt][nt][r];
                }
            }
        }
    }
}

// ---------------------------------------------------------------------------
// 4) Shared expert over all tokens (dense, writes out)
// ---------------------------------------------------------------------------
__global__ __launch_bounds__(512) void swiglu_shared_kernel(const __bf16* __restrict__ xb,
                                                            const __bf16* __restrict__ WuT,
                                                            const __bf16* __restrict__ WgT,
                                                            const __bf16* __restrict__ WdT,
                                                            float* __restrict__ out) {
    swiglu_tile<false>(xb, WuT, WgT, WdT, out, nullptr, nullptr, blockIdx.x * TM, TM);
}

// ---------------------------------------------------------------------------
// 5) Routed experts (gathered token tiles, accumulates scaled into out)
// ---------------------------------------------------------------------------
__global__ __launch_bounds__(512) void swiglu_routed_kernel(const __bf16* __restrict__ xb,
                                                            const __bf16* __restrict__ WuT,
                                                            const __bf16* __restrict__ WgT,
                                                            const __bf16* __restrict__ WdT,
                                                            const int* __restrict__ counts,
                                                            const int* __restrict__ lists,
                                                            const float* __restrict__ pvals,
                                                            float* __restrict__ out) {
    int e   = blockIdx.y;
    int cnt = counts[e];
    int t0  = blockIdx.x * TM;
    if (t0 >= cnt) return;
    int valid = cnt - t0; if (valid > TM) valid = TM;
    size_t wsz = (size_t)HID * DIM;
    swiglu_tile<true>(xb,
                      WuT + (size_t)e * wsz,
                      WgT + (size_t)e * wsz,
                      WdT + (size_t)e * wsz,
                      out,
                      lists + (size_t)e * NTOK + t0,
                      pvals, 0, valid);
}

// ---------------------------------------------------------------------------
// Host launcher
// ---------------------------------------------------------------------------
extern "C" void kernel_launch(void* const* d_in, const int* in_sizes, int n_in,
                              void* d_out, int out_size, void* d_ws, size_t ws_size,
                              hipStream_t stream) {
    const float* x   = (const float*)d_in[0];
    const float* Wr  = (const float*)d_in[1];
    const float* Wu  = (const float*)d_in[2];
    const float* Wg  = (const float*)d_in[3];
    const float* Wd  = (const float*)d_in[4];
    const float* Wus = (const float*)d_in[5];
    const float* Wgs = (const float*)d_in[6];
    const float* Wds = (const float*)d_in[7];
    float* out = (float*)d_out;

    // ---- workspace carve (bf16 weights/activations + router state) ----
    char* p = (char*)d_ws;
    const size_t wsz = (size_t)HID * DIM;              // elements per weight matrix
    __bf16* xb   = (__bf16*)p;  p += (size_t)NTOK * DIM * 2;
    __bf16* WuT  = (__bf16*)p;  p += (size_t)NREAL * wsz * 2;
    __bf16* WgT  = (__bf16*)p;  p += (size_t)NREAL * wsz * 2;
    __bf16* WdT  = (__bf16*)p;  p += (size_t)NREAL * wsz * 2;
    __bf16* WuTs = (__bf16*)p;  p += wsz * 2;
    __bf16* WgTs = (__bf16*)p;  p += wsz * 2;
    __bf16* WdTs = (__bf16*)p;  p += wsz * 2;
    float*  pvals  = (float*)p; p += (size_t)NTOK * 4;
    int*    counts = (int*)p;   p += 64;
    int*    lists  = (int*)p;   p += (size_t)NREAL * NTOK * 4;

    hipMemsetAsync(counts, 0, NEXP * sizeof(int), stream);

    // ---- pre-pass: bf16 convert + transpose ----
    cvt_bf16_kernel<<<4096, 256, 0, stream>>>(x, xb, NTOK * DIM);
    dim3 tb(32, 8);
    // Wu/Wg: [DIM x HID] -> T [HID x DIM]
    transpose_bf16_kernel<<<dim3(HID / 32, DIM / 32, NREAL), tb, 0, stream>>>(Wu, WuT, DIM, HID);
    transpose_bf16_kernel<<<dim3(HID / 32, DIM / 32, NREAL), tb, 0, stream>>>(Wg, WgT, DIM, HID);
    // Wd: [HID x DIM] -> T [DIM x HID]
    transpose_bf16_kernel<<<dim3(DIM / 32, HID / 32, NREAL), tb, 0, stream>>>(Wd, WdT, HID, DIM);
    transpose_bf16_kernel<<<dim3(HID / 32, DIM / 32, 1), tb, 0, stream>>>(Wus, WuTs, DIM, HID);
    transpose_bf16_kernel<<<dim3(HID / 32, DIM / 32, 1), tb, 0, stream>>>(Wgs, WgTs, DIM, HID);
    transpose_bf16_kernel<<<dim3(DIM / 32, HID / 32, 1), tb, 0, stream>>>(Wds, WdTs, HID, DIM);

    // ---- router (one wave32 per token) ----
    router_kernel<<<NTOK / 8, 256, 0, stream>>>(x, Wr, pvals, counts, lists);

    // ---- shared expert (writes out), then routed experts (accumulate) ----
    swiglu_shared_kernel<<<NTOK / TM, 512, 0, stream>>>(xb, WuTs, WgTs, WdTs, out);
    swiglu_routed_kernel<<<dim3(NTOK / TM, NREAL), 512, 0, stream>>>(
        xb, WuT, WgT, WdT, counts, lists, pvals, out);
}
